// MultiQueryAttention_56272661512447
// MI455X (gfx1250) — matrix operand order repaired and checked by reading, not compiled, not run on hardware
//
#include <hip/hip_runtime.h>
#include <hip/hip_bf16.h>

// ---------------------------------------------------------------------------
// MQA forward for MI455X (gfx1250, wave32, WMMA 16x16x32 f16, f32 accum).
//   1) convert hidden f32->f16 (Xh), transpose+convert weights to [N][K] f16
//   2) Q = (Xh@Wq + bq) * 0.125   -> Qh  [4096][1024] f16   (TDM-staged GEMM)
//   3) K =  Xh@Wk + bk            -> Kh  [4096][64]   f16
//   4) V =  Xh@Wv + bv (transp.)  -> Vt  [2][64][2048] f16
//   5) fused flash attention (software-pipelined)         -> Ctx f16
//   6) out = Ctx@Wo + bo (f32)    -> d_out               (TDM-staged GEMM)
// ---------------------------------------------------------------------------

typedef _Float16 v16h  __attribute__((ext_vector_type(16)));
typedef _Float16 half8 __attribute__((ext_vector_type(8)));
typedef float    v8f   __attribute__((ext_vector_type(8)));
typedef unsigned int u32x4 __attribute__((ext_vector_type(4)));
typedef int          i32x8 __attribute__((ext_vector_type(8)));
typedef int          i32x4 __attribute__((ext_vector_type(4)));

#define HIDDEN 1024
#define NHEADS 16
#define HDIM   64
#define SEQ    2048
#define BATCH  2
#define MROWS  (BATCH * SEQ) // 4096

union V16U { v16h v; half8 h[2]; };

// A-fragment (16x32 f16, M = lane%16): lane-half hh holds K = [16hh,16hh+16)
// -> two contiguous 16B loads from a K-major row.
__device__ __forceinline__ v16h load_a_frag(const _Float16* rowp, int hh) {
  V16U u;
  u.h[0] = *(const half8*)(rowp + 16 * hh);
  u.h[1] = *(const half8*)(rowp + 16 * hh + 8);
  return u.v;
}
// B-fragment (32x16 f16, N = lane%16): lane-half hh holds K = [16hh,16hh+16)
// -> one contiguous 32B load from a K-major (B-transposed) row.
__device__ __forceinline__ v16h load_b_frag(const _Float16* p) {
  V16U u;
  u.h[0] = *(const half8*)(p);
  u.h[1] = *(const half8*)(p + 8);
  return u.v;
}
__device__ __forceinline__ v8f wmma_f16(v16h a, v16h b, v8f c) {
  return __builtin_amdgcn_wmma_f32_16x16x32_f16(false, a, false, b, (short)0, c,
                                                false, false);
}

// ---------------------------------------------------------------------------
// TDM: issue a 2D tile load (global -> LDS) via tensor descriptor (ISA ch 8).
// All units are data_size elements (2B). LDS receives [tile1][tile0] packed.
// clang-23 6-arg builtin: (u32x4 g0, i32x8 g1, i32x4 g2, i32x4 g3, i32x8 g4, cpol)
// groups 2/3/4 only needed for 3D-5D tensors -> zero-filled.
__device__ __forceinline__ void tdm_load_2d(unsigned int lds_byte_addr,
                                            const void* gptr,
                                            unsigned int tdim0, unsigned int tdim1,
                                            unsigned int tile0, unsigned int tile1,
                                            unsigned long long stride0) {
  unsigned long long ga = (unsigned long long)(uintptr_t)gptr;
  u32x4 g0;
  g0.x = 1u;                                             // count=1 (valid user D#)
  g0.y = lds_byte_addr;                                  // bits 63:32 lds_addr
  g0.z = (unsigned int)ga;                               // bits 95:64 gaddr lo
  g0.w = (unsigned int)((ga >> 32) & 0x1FFFFFFu) | (2u << 30); // gaddr hi | type=2
  i32x8 g1;
  g1[0] = (int)(1u << 16);                               // data_size=1 -> 2 bytes
  g1[1] = (int)((tdim0 & 0xFFFFu) << 16);                // tensor_dim0[15:0]
  g1[2] = (int)((tdim0 >> 16) | ((tdim1 & 0xFFFFu) << 16));
  g1[3] = (int)((tdim1 >> 16) | (tile0 << 16));          // tile_dim0
  g1[4] = (int)(tile1 & 0xFFFFu);                        // tile_dim1 (tile_dim2=0)
  g1[5] = (int)(unsigned int)stride0;                    // tensor_dim0_stride lo
  g1[6] = (int)(unsigned int)((stride0 >> 32) & 0xFFFFu);
  g1[7] = 0;
  i32x4 z4 = {0, 0, 0, 0};
  i32x8 z8 = {0, 0, 0, 0, 0, 0, 0, 0};
  __builtin_amdgcn_tensor_load_to_lds(g0, g1, z4, z4, z8, 0);
}

// ---------------------------------------------------------------------------
__global__ void k_cvt(const float* __restrict__ s, _Float16* __restrict__ d, int n) {
  int i = blockIdx.x * blockDim.x + threadIdx.x;
  if (i < n) d[i] = (_Float16)s[i];
}

// src [K][N] f32 -> dst [N][K] f16
__global__ void k_transpose(const float* __restrict__ s, _Float16* __restrict__ d,
                            int K, int N) {
  int i = blockIdx.x * blockDim.x + threadIdx.x;
  if (i < K * N) {
    int k = i / N, n = i % N;
    d[(size_t)n * K + k] = (_Float16)s[i];
  }
}

// ---------------------------------------------------------------------------
// Small-N GEMM (K/V projections, N=64): one wave per 16x64 tile, operands
// streamed straight from L2, software pipelined: step k+1's fragments are
// issued before step k's WMMAs so s_wait_loadcnt overlaps the MACs.
// mode 0: f16 row-major; mode 1: f16 transposed Vt.
__global__ void k_gemm(const _Float16* __restrict__ A, const _Float16* __restrict__ BT,
                       const float* __restrict__ bias, float alpha,
                       int M, int N, int K,
                       _Float16* __restrict__ out16, int mode, int ldo, int Sdim) {
  const int lane = threadIdx.x & 31;
  const int wv   = threadIdx.x >> 5;
  const int ntn  = N >> 6;
  const int tile = blockIdx.x * (blockDim.x >> 5) + wv;
  if (tile >= (M >> 4) * ntn) return;
  const int mt = tile / ntn, nt = tile % ntn;
  const int m0 = mt << 4, n0 = nt << 6;
  const int hh = lane >> 4, ln = lane & 15;

  v8f acc[4] = {};
  const _Float16* arow = A + (size_t)(m0 + ln) * K;
  const _Float16* brow[4];
#pragma unroll
  for (int t = 0; t < 4; ++t) brow[t] = BT + (size_t)(n0 + 16 * t + ln) * K + 16 * hh;

  // prologue: fragments for k0 = 0
  v16h af = load_a_frag(arow, hh);
  v16h bf[4];
#pragma unroll
  for (int t = 0; t < 4; ++t) bf[t] = load_b_frag(brow[t]);

  for (int k0 = 0; k0 < K; k0 += 32) {
    const int kn = (k0 + 32 < K) ? (k0 + 32) : k0; // clamped: tail loads unused
    // issue next step's loads first ...
    v16h afn = load_a_frag(arow + kn, hh);
    v16h bfn[4];
#pragma unroll
    for (int t = 0; t < 4; ++t) bfn[t] = load_b_frag(brow[t] + kn);
    // ... then consume the current fragments
#pragma unroll
    for (int t = 0; t < 4; ++t) acc[t] = wmma_f16(af, bf[t], acc[t]);
    af = afn;
#pragma unroll
    for (int t = 0; t < 4; ++t) bf[t] = bfn[t];
  }

#pragma unroll
  for (int t = 0; t < 4; ++t) {
    const int col = n0 + 16 * t + ln;
    const float bv = bias ? bias[col] : 0.0f;
#pragma unroll
    for (int r = 0; r < 8; ++r) {
      const int row = m0 + r + 8 * hh;
      const float v = (acc[t][r] + bv) * alpha;
      if (mode == 0)
        out16[(size_t)row * ldo + col] = (_Float16)v;
      else
        out16[(size_t)(row / Sdim) * HDIM * Sdim + (size_t)col * Sdim + (row % Sdim)] =
            (_Float16)v;
    }
  }
}

// ---------------------------------------------------------------------------
// Big GEMM via TDM double buffering. Workgroup (8 waves) computes 64x256;
// each wave a 32x64 tile (8 f32 accum frags). Per 32-wide k-step wave 0 issues
// tensor_load_to_lds for A[64][32] + BT[256][32] of the NEXT step, so the DMA
// overlaps this step's 8 WMMAs; operands are read from LDS (ds_load_b128).
__global__ void k_gemm_tdm(const _Float16* __restrict__ A, const _Float16* __restrict__ BT,
                           const float* __restrict__ bias, float alpha,
                           int M, int N, int K,
                           _Float16* __restrict__ out16, float* __restrict__ out32,
                           int mode, int ldo) {
  __shared__ _Float16 sA[2][64 * 32];   //  8 KB
  __shared__ _Float16 sB[2][256 * 32];  // 32 KB
  const int lane = threadIdx.x & 31;
  const int wv   = threadIdx.x >> 5;
  const int hh   = lane >> 4, ln = lane & 15;
  const int ntn  = N >> 8;                 // tiles of 256 cols
  const int mt   = blockIdx.x / ntn;
  const int nt   = blockIdx.x % ntn;
  const int m0   = mt * 64, n0 = nt * 256;
  const int wm   = wv & 1, wn = wv >> 1;   // 2x4 wave grid

  v8f acc[2][4] = {};

  const unsigned int ldsA[2] = {(unsigned int)(uintptr_t)&sA[0][0],
                                (unsigned int)(uintptr_t)&sA[1][0]};
  const unsigned int ldsB[2] = {(unsigned int)(uintptr_t)&sB[0][0],
                                (unsigned int)(uintptr_t)&sB[1][0]};

  if (wv == 0) {
    tdm_load_2d(ldsA[0], A + (size_t)m0 * K, K, M, 32, 64, K);
    tdm_load_2d(ldsB[0], BT + (size_t)n0 * K, K, N, 32, 256, K);
  }

  int cur = 0;
  for (int k0 = 0; k0 < K; k0 += 32) {
    if (wv == 0) __builtin_amdgcn_s_wait_tensorcnt(0);
    __syncthreads(); // buf[cur] ready; all reads of buf[cur^1] (prev step) done
    if (wv == 0 && k0 + 32 < K) {
      tdm_load_2d(ldsA[cur ^ 1], A + (size_t)m0 * K + (k0 + 32), K, M, 32, 64, K);
      tdm_load_2d(ldsB[cur ^ 1], BT + (size_t)n0 * K + (k0 + 32), K, N, 32, 256, K);
    }
    v16h af0 = load_a_frag(&sA[cur][(wm * 32 + ln) * 32], hh);
    v16h af1 = load_a_frag(&sA[cur][(wm * 32 + 16 + ln) * 32], hh);
#pragma unroll
    for (int t = 0; t < 4; ++t) {
      v16h bf = load_b_frag(&sB[cur][(wn * 64 + 16 * t + ln) * 32 + 16 * hh]);
      acc[0][t] = wmma_f16(af0, bf, acc[0][t]);
      acc[1][t] = wmma_f16(af1, bf, acc[1][t]);
    }
    cur ^= 1;
  }

#pragma unroll
  for (int a = 0; a < 2; ++a)
#pragma unroll
    for (int t = 0; t < 4; ++t) {
      const int col = n0 + wn * 64 + 16 * t + ln;
      const float bv = bias ? bias[col] : 0.0f;
#pragma unroll
      for (int r = 0; r < 8; ++r) {
        const int row = m0 + wm * 32 + a * 16 + r + 8 * hh;
        const float v = (acc[a][t][r] + bv) * alpha;
        if (mode == 0)
          out16[(size_t)row * ldo + col] = (_Float16)v;
        else
          out32[(size_t)row * ldo + col] = v;
      }
    }
}

// ---------------------------------------------------------------------------
// Fused flash attention, one wave per (batch, head, 16-row q tile).
// Software pipelined: V frags for step i and K frags for step i+1 are issued
// before the softmax VALU/shfl/LDS work so their latency is hidden.
__global__ void k_attn(const _Float16* __restrict__ Qh, const _Float16* __restrict__ Kh,
                       const _Float16* __restrict__ Vt, const int* __restrict__ mask,
                       _Float16* __restrict__ Ctx) {
  __shared__ _Float16 lds[8][16][32]; // per-wave P tile (16 q x 32 sk)
  const int lane = threadIdx.x & 31;
  const int wv   = threadIdx.x >> 5;
  const int hh   = lane >> 4, ln = lane & 15;

  const int tile      = blockIdx.x * 8 + wv;
  const int qtPerHead = SEQ / 16;
  const int b   = tile / (NHEADS * qtPerHead);
  const int rem = tile % (NHEADS * qtPerHead);
  const int h   = rem / qtPerHead;
  const int q0  = (rem % qtPerHead) * 16;

  const _Float16* qrow = Qh + (size_t)(b * SEQ + q0 + ln) * HIDDEN + h * HDIM;
  const v16h aq0 = load_a_frag(qrow, hh);
  const v16h aq1 = load_a_frag(qrow + 32, hh);

  v8f ctx[4] = {};
  float mst[8], lst[8];
#pragma unroll
  for (int r = 0; r < 8; ++r) { mst[r] = -3.0e38f; lst[r] = 0.0f; }

  const _Float16* kbase = Kh + (size_t)b * SEQ * HDIM;
  const _Float16* vbase = Vt + (size_t)b * HDIM * SEQ;
  const int* mrow = mask + b * SEQ;

  // preload K fragments for sk0 = 0
  v16h kf[4];
  {
    const _Float16* kr0 = kbase + (size_t)ln * HDIM + 16 * hh;
    const _Float16* kr1 = kbase + (size_t)(16 + ln) * HDIM + 16 * hh;
    kf[0] = load_b_frag(kr0);
    kf[1] = load_b_frag(kr0 + 32);
    kf[2] = load_b_frag(kr1);
    kf[3] = load_b_frag(kr1 + 32);
  }

  for (int sk0 = 0; sk0 < SEQ; sk0 += 32) {
    // ---- scores (consume preloaded K frags)
    v8f s0 = {}, s1 = {};
    s0 = wmma_f16(aq0, kf[0], s0);
    s0 = wmma_f16(aq1, kf[1], s0);
    s1 = wmma_f16(aq0, kf[2], s1);
    s1 = wmma_f16(aq1, kf[3], s1);

    // ---- issue V frags (used only after softmax) and next K frags now
    v16h vf[4];
#pragma unroll
    for (int t = 0; t < 4; ++t)
      vf[t] = load_b_frag(vbase + (size_t)(16 * t + ln) * SEQ + sk0 + 16 * hh);
    if (sk0 + 32 < SEQ) {
      const _Float16* kr0 = kbase + (size_t)(sk0 + 32 + ln) * HDIM + 16 * hh;
      const _Float16* kr1 = kbase + (size_t)(sk0 + 48 + ln) * HDIM + 16 * hh;
      kf[0] = load_b_frag(kr0);
      kf[1] = load_b_frag(kr0 + 32);
      kf[2] = load_b_frag(kr1);
      kf[3] = load_b_frag(kr1 + 32);
    }
    if (sk0 + 64 < SEQ) // warm L2->L0 path two steps ahead
      __builtin_prefetch(kbase + (size_t)(sk0 + 64 + ln) * HDIM, 0, 0);

    const float madd0 = (mrow[sk0 + ln] != 0) ? 0.0f : -3.0e38f;
    const float madd1 = (mrow[sk0 + 16 + ln] != 0) ? 0.0f : -3.0e38f;
#pragma unroll
    for (int r = 0; r < 8; ++r) { s0[r] += madd0; s1[r] += madd1; }

    // ---- online softmax (row stats replicated across each 16-lane half)
    float corr[8];
#pragma unroll
    for (int r = 0; r < 8; ++r) {
      float rm = fmaxf(s0[r], s1[r]);
#pragma unroll
      for (int off = 1; off < 16; off <<= 1)
        rm = fmaxf(rm, __shfl_xor(rm, off, 32));
      const float nm = fmaxf(mst[r], rm);
      corr[r] = __expf(mst[r] - nm);
      s0[r]   = __expf(s0[r] - nm);
      s1[r]   = __expf(s1[r] - nm);
      float rs = s0[r] + s1[r];
#pragma unroll
      for (int off = 1; off < 16; off <<= 1)
        rs += __shfl_xor(rs, off, 32);
      lst[r] = lst[r] * corr[r] + rs;
      mst[r] = nm;
    }
#pragma unroll
    for (int t = 0; t < 4; ++t)
#pragma unroll
      for (int r = 0; r < 8; ++r) ctx[t][r] *= corr[r];

    // ---- P: C-layout -> A-layout via per-wave LDS tile
#pragma unroll
    for (int r = 0; r < 8; ++r) {
      lds[wv][r + 8 * hh][ln]      = (_Float16)s0[r];
      lds[wv][r + 8 * hh][16 + ln] = (_Float16)s1[r];
    }
    asm volatile("s_wait_dscnt 0" ::: "memory"); // intra-wave DS RAW
    const v16h pA = load_a_frag(&lds[wv][ln][0], hh);

    // ---- ctx += P @ V
#pragma unroll
    for (int t = 0; t < 4; ++t) ctx[t] = wmma_f16(pA, vf[t], ctx[t]);
  }

  float inv[8];
#pragma unroll
  for (int r = 0; r < 8; ++r) inv[r] = 1.0f / lst[r];
#pragma unroll
  for (int t = 0; t < 4; ++t)
#pragma unroll
    for (int r = 0; r < 8; ++r) {
      const int row = q0 + r + 8 * hh;
      Ctx[(size_t)(b * SEQ + row) * HIDDEN + h * HDIM + 16 * t + ln] =
          (_Float16)(ctx[t][r] * inv[r]);
    }
}

// ---------------------------------------------------------------------------
extern "C" void kernel_launch(void* const* d_in, const int* in_sizes, int n_in,
                              void* d_out, int out_size, void* d_ws, size_t ws_size,
                              hipStream_t stream) {
  const float* hidden = (const float*)d_in[0];
  const int*   amask  = (const int*)d_in[1];
  const float* Wq = (const float*)d_in[2];
  const float* bq = (const float*)d_in[3];
  const float* Wk = (const float*)d_in[4];
  const float* bk = (const float*)d_in[5];
  const float* Wv = (const float*)d_in[6];
  const float* bv = (const float*)d_in[7];
  const float* Wo = (const float*)d_in[8];
  const float* bo = (const float*)d_in[9];
  float* out = (float*)d_out;

  char* ws = (char*)d_ws; // ~29.3 MB used
  _Float16* Xh  = (_Float16*)(ws);
  _Float16* Qh  = (_Float16*)(ws + (8u << 20));
  _Float16* Ctx = (_Float16*)(ws + (16u << 20));
  _Float16* WqT = (_Float16*)(ws + (24u << 20));
  _Float16* WoT = (_Float16*)(ws + (26u << 20));
  _Float16* WkT = (_Float16*)(ws + (28u << 20));
  _Float16* WvT = (_Float16*)(ws + (28u << 20) + (1u << 17));
  _Float16* Kh  = (_Float16*)(ws + (28u << 20) + (2u << 17));
  _Float16* Vt  = (_Float16*)(ws + (28u << 20) + (6u << 17));

  const int nX = MROWS * HIDDEN;
  k_cvt<<<nX / 256, 256, 0, stream>>>(hidden, Xh, nX);
  k_transpose<<<(HIDDEN * HIDDEN) / 256, 256, 0, stream>>>(Wq, WqT, HIDDEN, HIDDEN);
  k_transpose<<<(HIDDEN * HDIM) / 256, 256, 0, stream>>>(Wk, WkT, HIDDEN, HDIM);
  k_transpose<<<(HIDDEN * HDIM) / 256, 256, 0, stream>>>(Wv, WvT, HIDDEN, HDIM);
  k_transpose<<<(HIDDEN * HIDDEN) / 256, 256, 0, stream>>>(Wo, WoT, HIDDEN, HIDDEN);

  // Q projection (alpha folds 1/sqrt(64)); TDM-staged GEMM, 64x256 per block
  k_gemm_tdm<<<(MROWS / 64) * (HIDDEN / 256), 256, 0, stream>>>(
      Xh, WqT, bq, 0.125f, MROWS, HIDDEN, HIDDEN, Qh, nullptr, 0, HIDDEN);
  // K / V projections (N=64): pipelined direct-global per-wave GEMM
  k_gemm<<<(MROWS / 16) * (HDIM / 64) / 8, 256, 0, stream>>>(
      Xh, WkT, bk, 1.0f, MROWS, HDIM, HIDDEN, Kh, 0, HDIM, SEQ);
  k_gemm<<<(MROWS / 16) * (HDIM / 64) / 8, 256, 0, stream>>>(
      Xh, WvT, bv, 1.0f, MROWS, HDIM, HIDDEN, Vt, 1, 0, SEQ);

  // fused attention
  k_attn<<<(BATCH * NHEADS * (SEQ / 16)) / 8, 256, 0, stream>>>(Qh, Kh, Vt, amask, Ctx);

  // output projection, f32 result; TDM-staged GEMM
  k_gemm_tdm<<<(MROWS / 64) * (HIDDEN / 256), 256, 0, stream>>>(
      Ctx, WoT, bo, 1.0f, MROWS, HIDDEN, HIDDEN, nullptr, out, 2, HIDDEN);
}
